// FCOS_72739566125891
// MI455X (gfx1250) — matrix operand order repaired and checked by reading, not compiled, not run on hardware
//
#include <hip/hip_runtime.h>
#include <hip/hip_bf16.h>

// ---------------------------------------------------------------------------
// FCOS head forward for MI455X (gfx1250).
// Tower convs: implicit-GEMM conv3x3, A tile staged to LDS with
//   global_load_async_to_lds_b128 (ASYNCcnt) and consumed via ds_load_b128,
//   v_wmma_f32_16x16x32_bf16 with A reuse across 16 cout tiles / 4 waves.
// Heads: direct-global NT=1 WMMA kernel with zero-page boundary handling.
// GroupNorm stats fused into the tower conv epilogue (butterfly + f32 atomics).
// ---------------------------------------------------------------------------

typedef __attribute__((ext_vector_type(16))) __bf16        v16bf;
typedef __attribute__((ext_vector_type(8)))  float         v8f;
typedef __attribute__((ext_vector_type(4)))  unsigned int  u32x4;

union FragU { v16bf v; u32x4 u[2]; };

#define CIN       256
#define KDIM      2304      // 9 taps * 256 cin
#define NCHUNK    72        // KDIM / 32
#define GROUPS    16
#define TOTPOS    17064     // sum of H*W over 5 levels
#define NBATCH    2

// ---------------------------------------------------------------------------
// NCHW fp32 -> per-pixel-channel bf16 ([p][c], p = n*HW + h*W + w)
// ---------------------------------------------------------------------------
__global__ void nchw_to_pc_bf16(const float* __restrict__ x,
                                __hip_bfloat16* __restrict__ y,
                                int HW, int totalP) {
    int i = blockIdx.x * 256 + threadIdx.x;
    if (i >= totalP * CIN) return;
    int c = i & 255;
    int p = i >> 8;
    int n = p / HW, pos = p % HW;
    y[i] = __float2bfloat16(x[((size_t)n * CIN + c) * HW + pos]);
}

// ---------------------------------------------------------------------------
// OIHW fp32 weights -> bf16 [coutPad][K], K = tap*256 + cin, zero-padded rows
// ---------------------------------------------------------------------------
__global__ void pack_weights_bf16(const float* __restrict__ src,
                                  __hip_bfloat16* __restrict__ dst,
                                  int coutTotal, int coutPad) {
    int i = blockIdx.x * 256 + threadIdx.x;
    if (i >= coutPad * KDIM) return;
    int cout = i / KDIM;
    int k = i - cout * KDIM;
    int tap = k >> 8;        // 0..8
    int ci  = k & 255;
    float v = (cout < coutTotal) ? src[((size_t)cout * CIN + ci) * 9 + tap] : 0.0f;
    dst[i] = __float2bfloat16(v);
}

// ---------------------------------------------------------------------------
// Tower conv3x3 (256->256) + bias + GroupNorm partial stats.
// Block: 128 threads (4 waves) per 16-pixel tile; wave wv covers cout tiles
// wv*4 .. wv*4+3 -> all 256 couts per block.
// Phase 1: all threads stage the full A tile (72 chunks x 1KB, fragment
//          layout) into LDS with global_load_async_to_lds_b128; boundary taps
//          read a zeroed page. s_wait_asynccnt + barrier.
// Phase 2: per wave, 72 chunks: A from LDS (2x ds_load_b128),
//          4x (2x global B load + v_wmma_f32_16x16x32_bf16).
// ---------------------------------------------------------------------------
__global__ __launch_bounds__(128)
void conv3x3_tower_async(const __hip_bfloat16* __restrict__ xbf,
                         const __hip_bfloat16* __restrict__ wbf,
                         const __hip_bfloat16* __restrict__ zeropage,
                         const float* __restrict__ bias,
                         int H, int W, int HW, int totalP,
                         float* __restrict__ z,
                         float* __restrict__ stat_sum,
                         float* __restrict__ stat_ssq) {
    __shared__ __align__(16) char sA[NCHUNK * 1024];   // 72 KB A staging

    const int tile = blockIdx.x;
    const int tid  = threadIdx.x;
    const int wv   = tid >> 5;         // wave 0..3
    const int l    = tid & 31;         // lane
    const int half = l >> 4;
    const int lm   = l & 15;

    // ---- Phase 1: async-stage A tile into LDS in fragment layout ----------
    // chunk k (tap=k>>3, cc=k&7): 1KB = 32 lanes x {u0,u1} x 16B
    //   u0 at k*1024 + lane*16, u1 at k*1024 + 512 + lane*16
    const unsigned sbase = (unsigned)(size_t)(void*)sA;  // LDS aperture: low 32b = offset
    for (int idx = tid; idx < NCHUNK * 64; idx += 128) {
        const int k     = idx >> 6;
        const int sub   = idx & 63;
        const int lane  = sub & 31;
        const int which = sub >> 5;
        const int tap   = k >> 3;
        const int cc    = k & 7;
        const int m     = lane & 15;
        const int hl    = lane >> 4;
        const int cin   = cc * 32 + hl * 16 + which * 8;

        const int  pp  = tile * 16 + m;
        const bool pvv = pp < totalP;
        const int  ppc = pvv ? pp : 0;
        const int  nn  = ppc / HW;
        const int  pos = ppc - nn * HW;
        const int  hh  = pos / W + (tap / 3) - 1;
        const int  ww  = pos - (pos / W) * W + (tap % 3) - 1;
        const bool av  = pvv && ((unsigned)hh < (unsigned)H) && ((unsigned)ww < (unsigned)W);
        const int  disp = ((tap / 3) - 1) * W + ((tap % 3) - 1);

        const __hip_bfloat16* g = av ? (xbf + ((size_t)ppc + disp) * CIN + cin) : zeropage;
        const unsigned ldsoff = sbase + (unsigned)(k * 1024 + which * 512 + lane * 16);
        const unsigned long long ga = (unsigned long long)(size_t)g;
        asm volatile("global_load_async_to_lds_b128 %0, %1, off"
                     :: "v"(ldsoff), "v"(ga) : "memory");
    }
    asm volatile("s_wait_asynccnt 0x0" ::: "memory");
    __syncthreads();

    // ---- Phase 2: WMMA main loop ------------------------------------------
    v8f acc[4];
    #pragma unroll
    for (int t = 0; t < 4; ++t)
        #pragma unroll
        for (int i = 0; i < 8; ++i) acc[t][i] = 0.0f;

    const u32x4* wp[4];
    #pragma unroll
    for (int t = 0; t < 4; ++t)
        wp[t] = (const u32x4*)(wbf + (size_t)((wv * 4 + t) * 16 + lm) * KDIM + half * 16);

    const u32x4* la0 = ((const u32x4*)sA) + l;   // + k*64 per chunk; u1 = +32

    #pragma unroll 1
    for (int k = 0; k < NCHUNK; ++k) {
        FragU A;
        A.u[0] = la0[k * 64];
        A.u[1] = la0[k * 64 + 32];
        #pragma unroll
        for (int t = 0; t < 4; ++t) {
            FragU B;
            B.u[0] = wp[t][k * 4];
            B.u[1] = wp[t][k * 4 + 1];
            acc[t] = __builtin_amdgcn_wmma_f32_16x16x32_bf16(
                false, A.v, false, B.v, (short)0, acc[t], false, false);
        }
    }

    // ---- Epilogue: bias, z store, GN partial sums -------------------------
    const int baseRow = tile * 16 + half * 8;
    const bool uni = ((tile * 16) / HW == (tile * 16 + 15) / HW) &&
                     (tile * 16 + 15 < totalP);
    #pragma unroll
    for (int t = 0; t < 4; ++t) {
        const int g  = wv * 4 + t;            // GN group == cout tile index
        const int cout = g * 16 + lm;
        const float bv = bias[cout];
        if (uni) {
            const int nn = (tile * 16) / HW;
            float s = 0.0f, s2 = 0.0f;
            #pragma unroll
            for (int r = 0; r < 8; ++r) {
                const int pr = baseRow + r;
                const float zv = acc[t][r] + bv;
                z[(size_t)pr * CIN + cout] = zv;
                s  += zv;
                s2 += zv * zv;
            }
            #pragma unroll
            for (int off = 16; off >= 1; off >>= 1) {
                s  += __shfl_xor(s,  off, 32);
                s2 += __shfl_xor(s2, off, 32);
            }
            if (l == 0) {
                atomicAdd(&stat_sum[nn * GROUPS + g], s);
                atomicAdd(&stat_ssq[nn * GROUPS + g], s2);
            }
        } else {
            #pragma unroll
            for (int r = 0; r < 8; ++r) {
                const int pr = baseRow + r;
                if (pr < totalP) {
                    const float zv = acc[t][r] + bv;
                    z[(size_t)pr * CIN + cout] = zv;
                    atomicAdd(&stat_sum[(pr / HW) * GROUPS + g], zv);
                    atomicAdd(&stat_ssq[(pr / HW) * GROUPS + g], zv * zv);
                }
            }
        }
    }
}

// ---------------------------------------------------------------------------
// Head conv3x3 (256 -> coutTotal<=16*gridDim.y), one wave per tile,
// direct-global A with zero-page select, writes fp32 to d_out layout
// [(n*TOTPOS + levelOff + pos)*stride + cout], optional ReLU.
// ---------------------------------------------------------------------------
__global__ __launch_bounds__(32)
void conv3x3_head_wmma(const __hip_bfloat16* __restrict__ xbf,
                       const __hip_bfloat16* __restrict__ wbf,
                       const __hip_bfloat16* __restrict__ zeropage,
                       const float* __restrict__ bias,
                       int H, int W, int HW, int totalP,
                       float* __restrict__ outp,
                       int outStride, int coutTotal, int reluFlag,
                       int levelOff) {
    const int tile = blockIdx.x;
    const int ct   = blockIdx.y;
    const int l    = threadIdx.x;
    const int half = l >> 4;
    const int lm   = l & 15;

    const int  p   = tile * 16 + lm;
    const bool pv  = p < totalP;
    const int  pc  = pv ? p : 0;
    const int  n   = pc / HW;
    const int  pos = pc - n * HW;
    const int  h   = pos / W;
    const int  w   = pos - h * W;

    v8f acc;
    #pragma unroll
    for (int i = 0; i < 8; ++i) acc[i] = 0.0f;

    const u32x4* centre = ((const u32x4*)(xbf + (size_t)pc * CIN)) + half;
    const u32x4* zp4    = ((const u32x4*)zeropage) + half;
    const u32x4* wp = (const u32x4*)(wbf + (size_t)(ct * 16 + lm) * KDIM + half * 8);

    #pragma unroll 1
    for (int tap = 0; tap < 9; ++tap) {
        const int dh = tap / 3 - 1;
        const int dw = tap % 3 - 1;
        const int hh = h + dh;
        const int ww = w + dw;
        const bool av = pv && ((unsigned)hh < (unsigned)H) && ((unsigned)ww < (unsigned)W);
        const int doff = (dh * W + dw) * (CIN / 8);     // wave-uniform
        const u32x4* ap = av ? (centre + doff) : zp4;
        #pragma unroll
        for (int cc = 0; cc < 8; ++cc) {
            FragU A, B;
            A.u[0] = ap[cc * 4];
            A.u[1] = ap[cc * 4 + 2];
            const int kc = tap * 8 + cc;
            B.u[0] = wp[kc * 4];
            B.u[1] = wp[kc * 4 + 2];
            acc = __builtin_amdgcn_wmma_f32_16x16x32_bf16(
                false, A.v, false, B.v, (short)0, acc, false, false);
        }
    }

    const int cout = ct * 16 + lm;
    if (cout < coutTotal) {
        const float bv = bias[cout];
        const int baseRow = tile * 16 + half * 8;
        #pragma unroll
        for (int r = 0; r < 8; ++r) {
            const int pr = baseRow + r;
            if (pr < totalP) {
                float v = acc[r] + bv;
                if (reluFlag) v = fmaxf(v, 0.0f);
                const int nr = pr / HW;
                const int posr = pr - nr * HW;
                outp[((size_t)nr * TOTPOS + levelOff + posr) * outStride + cout] = v;
            }
        }
    }
}

// ---------------------------------------------------------------------------
// GroupNorm normalize + affine + ReLU, fp32 z -> bf16 activation
// ---------------------------------------------------------------------------
__global__ void gn_relu_bf16(const float* __restrict__ z,
                             const float* __restrict__ ssum,
                             const float* __restrict__ ssq,
                             const float* __restrict__ gamma,
                             const float* __restrict__ beta,
                             __hip_bfloat16* __restrict__ y,
                             int HW, int totalP) {
    int i = blockIdx.x * 256 + threadIdx.x;
    if (i >= totalP * CIN) return;
    int c = i & 255;
    int p = i >> 8;
    int n = p / HW;
    int g = c >> 4;
    float cnt  = 16.0f * (float)HW;
    float mean = ssum[n * GROUPS + g] / cnt;
    float var  = ssq[n * GROUPS + g] / cnt - mean * mean;
    float v = (z[i] - mean) * rsqrtf(var + 1e-5f) * gamma[c] + beta[c];
    v = fmaxf(v, 0.0f);
    y[i] = __float2bfloat16(v);
}

// ---------------------------------------------------------------------------
// Host orchestration
// ---------------------------------------------------------------------------
extern "C" void kernel_launch(void* const* d_in, const int* in_sizes, int n_in,
                              void* d_out, int out_size, void* d_ws, size_t ws_size,
                              hipStream_t stream) {
    (void)in_sizes; (void)n_in; (void)out_size; (void)ws_size;

    // pytree order (dict keys sorted): feat0..4, bbox_reg(w,b),
    // cls_convs 3x(w,b,g,bt), cls_logits(w,b), ctrness(w,b), reg_convs 3x(w,b,g,bt)
    const float* feats[5] = {(const float*)d_in[0], (const float*)d_in[1],
                             (const float*)d_in[2], (const float*)d_in[3],
                             (const float*)d_in[4]};

    // workspace carve-out
    const size_t WROWS = 1648;                       // packed bf16 weight rows
    const size_t MAXP  = (size_t)NBATCH * 100 * 128; // 25600 pixels, level 0
    char* ws = (char*)d_ws;
    size_t off = 0;
    __hip_bfloat16* wbf = (__hip_bfloat16*)(ws + off); off += WROWS * KDIM * sizeof(__hip_bfloat16);
    off = (off + 255) & ~(size_t)255;
    __hip_bfloat16* zp = (__hip_bfloat16*)(ws + off);  off += 1024;  // zero page
    float* ssum = (float*)(ws + off);
    float* ssq  = ssum + NBATCH * GROUPS;              off += 256;
    float* zbuf = (float*)(ws + off);                  off += MAXP * CIN * sizeof(float);
    __hip_bfloat16* b0 = (__hip_bfloat16*)(ws + off);  off += MAXP * CIN * sizeof(__hip_bfloat16);
    __hip_bfloat16* b1 = (__hip_bfloat16*)(ws + off);  off += MAXP * CIN * sizeof(__hip_bfloat16);
    __hip_bfloat16* b2 = (__hip_bfloat16*)(ws + off);

    hipMemsetAsync(zp, 0, 1024, stream);   // zero page for OOB taps

    // --- pack all conv weights to bf16 [cout][K] ---
    struct Pack { int src; int rowOff; int coutTotal; int coutPad; };
    const Pack packs[9] = {
        { 7,    0, 256, 256}, {11,  256, 256, 256}, {15,  512, 256, 256},  // cls tower
        {23,  768, 256, 256}, {27, 1024, 256, 256}, {31, 1280, 256, 256},  // reg tower
        {19, 1536,  80,  80},                                              // cls_logits
        { 5, 1616,   4,  16},                                              // bbox_reg
        {21, 1632,   1,  16}                                               // ctrness
    };
    for (int i = 0; i < 9; ++i) {
        int tot = packs[i].coutPad * KDIM;
        pack_weights_bf16<<<(tot + 255) / 256, 256, 0, stream>>>(
            (const float*)d_in[packs[i].src], wbf + (size_t)packs[i].rowOff * KDIM,
            packs[i].coutTotal, packs[i].coutPad);
    }

    float* cls_out = (float*)d_out;
    float* reg_out = cls_out + (size_t)NBATCH * TOTPOS * 80;
    float* ctr_out = reg_out + (size_t)NBATCH * TOTPOS * 4;

    const int Hs[5]   = {100, 50, 25, 13, 7};
    const int Wsz[5]  = {128, 64, 32, 16, 8};
    const int offs[5] = {0, 12800, 16000, 16800, 17008};

    for (int li = 0; li < 5; ++li) {
        const int H = Hs[li], W = Wsz[li], HW = H * W;
        const int totalP = NBATCH * HW;
        const int nelem  = totalP * CIN;
        const int ptiles = (totalP + 15) / 16;

        nchw_to_pc_bf16<<<(nelem + 255) / 256, 256, 0, stream>>>(feats[li], b0, HW, totalP);

        auto tower = [&](int baseIdx, int rowBase) -> __hip_bfloat16* {
            __hip_bfloat16* cur = b0;
            __hip_bfloat16* outs[3] = {b1, b2, b1};
            for (int t = 0; t < 3; ++t) {
                hipMemsetAsync(ssum, 0, 2 * NBATCH * GROUPS * sizeof(float), stream);
                conv3x3_tower_async<<<dim3(ptiles), 128, 0, stream>>>(
                    cur, wbf + (size_t)(rowBase + t * 256) * KDIM, zp,
                    (const float*)d_in[baseIdx + t * 4 + 1],
                    H, W, HW, totalP, zbuf, ssum, ssq);
                gn_relu_bf16<<<(nelem + 255) / 256, 256, 0, stream>>>(
                    zbuf, ssum, ssq,
                    (const float*)d_in[baseIdx + t * 4 + 2],
                    (const float*)d_in[baseIdx + t * 4 + 3],
                    outs[t], HW, totalP);
                cur = outs[t];
            }
            return cur;  // == b1
        };

        // --- classification branch ---
        __hip_bfloat16* ctow = tower(7, 0);
        conv3x3_head_wmma<<<dim3(ptiles, 5), 32, 0, stream>>>(
            ctow, wbf + (size_t)1536 * KDIM, zp, (const float*)d_in[20],
            H, W, HW, totalP, cls_out, 80, 80, /*relu=*/0, offs[li]);

        // --- regression branch ---
        __hip_bfloat16* rtow = tower(23, 768);
        conv3x3_head_wmma<<<dim3(ptiles, 1), 32, 0, stream>>>(
            rtow, wbf + (size_t)1616 * KDIM, zp, (const float*)d_in[6],
            H, W, HW, totalP, reg_out, 4, 4, /*relu=*/1, offs[li]);
        conv3x3_head_wmma<<<dim3(ptiles, 1), 32, 0, stream>>>(
            rtow, wbf + (size_t)1632 * KDIM, zp, (const float*)d_in[22],
            H, W, HW, totalP, ctr_out, 1, 1, /*relu=*/0, offs[li]);
    }
}
